// MatchLSTM_52810917872337
// MI455X (gfx1250) — matrix operand order repaired and checked by reading, not compiled, not run on hardware
//
#include <hip/hip_runtime.h>

// ---------------- problem dims ----------------
#define TPp 400
#define TQq 60
#define BB  64
// H=150, 2H=300, 4H=600, E=300 ; padded: 160 / 320 / 640 / 320 ; 8H=1200 -> 1216

typedef __attribute__((ext_vector_type(16))) _Float16 v16h;
typedef __attribute__((ext_vector_type(8)))  float    v8f;

__device__ __forceinline__ float sigf(float x){ return 1.0f/(1.0f+expf(-x)); }

// ---------- WMMA fragment helpers (CDNA5 v_wmma_f32_16x16x32_f16) ----------
// A (16x32 f16, row-major src, leading dim lda): lane m=lane&15, half=lane>>4
//   halfword i: k = 16*(i>>3) + 8*half + (i&7)  -> two contiguous 16B chunks
__device__ __forceinline__ v16h frag_a(const _Float16* A, int lda, int row0, int k0, int lane){
  int m = lane & 15, hf = lane >> 4;
  const _Float16* p0 = A + (size_t)(row0 + m)*lda + (k0 + 8*hf);
  const _Float16* p1 = p0 + 16;
  v16h a;
#pragma unroll
  for (int j = 0; j < 8; ++j){ a[j] = p0[j]; a[j+8] = p1[j]; }
  return a;
}
// B (32x16 f16) built from weight W stored row-major [N][K] (B[k][n] = W[n][k]):
//   lane n=lane&15, half=lane>>4; halfword i: k = k0 + 16*(i>>3) + 8*half + (i&7) -> two 16B chunks
__device__ __forceinline__ v16h frag_b(const _Float16* W, int ldw, int n0, int k0, int lane){
  int n = lane & 15, hf = lane >> 4;
  const _Float16* p = W + (size_t)(n0 + n)*ldw + (k0 + 8*hf);
  v16h b;
#pragma unroll
  for (int j = 0; j < 8; ++j){ b[j] = p[j]; b[j+8] = p[j+16]; }
  return b;
}
// C/D f32 16x16: VGPR v -> row = v + 8*half, col = lane&15
__device__ __forceinline__ void frag_store(float* C, int ldc, int row0, int col0, int lane, v8f d){
  int n = lane & 15, hf = lane >> 4;
#pragma unroll
  for (int v = 0; v < 8; ++v) C[(size_t)(row0 + v + 8*hf)*ldc + col0 + n] = d[v];
}
#define WMMA(acc,a,b) (acc) = __builtin_amdgcn_wmma_f32_16x16x32_f16(false,(a),false,(b),(short)0,(acc),false,false)

// ---------------- prep kernels ----------------
__global__ void pad_w16_kernel(const float* __restrict__ src, _Float16* __restrict__ dst,
                               int N, int K, int ldsrc, int koff, int Npad, int Kpad){
  int idx = blockIdx.x*blockDim.x + threadIdx.x;
  int total = Npad*Kpad;
  if (idx >= total) return;
  int n = idx / Kpad, k = idx - n*Kpad;
  dst[idx] = (n < N && k < K) ? (_Float16)src[(size_t)n*ldsrc + koff + k] : (_Float16)0.0f;
}
__global__ void pad_bias_kernel(const float* __restrict__ a, const float* __restrict__ b,
                                float* __restrict__ dst, int N, int Npad){
  int idx = blockIdx.x*blockDim.x + threadIdx.x;
  if (idx >= Npad) return;
  float v = 0.0f;
  if (idx < N){ v = a[idx]; if (b) v += b[idx]; }
  dst[idx] = v;
}
__global__ void gather_emb_kernel(const int* __restrict__ ids, const float* __restrict__ emb,
                                  _Float16* __restrict__ out, int rows){
  int idx = blockIdx.x*blockDim.x + threadIdx.x;
  int total = rows*320;
  if (idx >= total) return;
  int r = idx / 320, k = idx - r*320;
  out[idx] = (k < 300) ? (_Float16)emb[(size_t)ids[r]*300 + k] : (_Float16)0.0f;
}
__global__ void zero_h16_kernel(_Float16* __restrict__ p, size_t n){
  size_t idx = (size_t)blockIdx.x*blockDim.x + threadIdx.x;
  if (idx < n) p[idx] = (_Float16)0.0f;
}

// ---------------- generic WMMA GEMM: C[M][N] = A_h[M][K] * W_h[N][K]^T + bias ----------------
__global__ __launch_bounds__(256) void gemm_f16f32(
    const _Float16* __restrict__ A, int lda,
    const _Float16* __restrict__ W, int ldw,
    const float* __restrict__ bias,
    float* __restrict__ C, int ldc,
    int Mt, int Nt, int K)
{
  int lane = threadIdx.x & 31;
  int wid  = (blockIdx.x*blockDim.x + threadIdx.x) >> 5;
  if (wid >= Mt*Nt) return;
  int tm = wid / Nt, tn = wid - tm*Nt;
  int row0 = tm*16, col0 = tn*16;
  v8f acc = {};
  for (int k0 = 0; k0 < K; k0 += 32){
    v16h a = frag_a(A, lda, row0, k0, lane);
    v16h b = frag_b(W, ldw, col0, k0, lane);
    WMMA(acc, a, b);
  }
  int n = lane & 15, hf = lane >> 4;
  float bv = bias ? bias[col0 + n] : 0.0f;
#pragma unroll
  for (int v = 0; v < 8; ++v)
    C[(size_t)(row0 + v + 8*hf)*ldc + col0 + n] = acc[v] + bv;
}

// ---------------- persistent bi-LSTM preprocess (4 blocks: pf,pb,qf,qb) ----------------
// Reference quirk: h is overwritten with tanh(c)*m, so the o-gate is dead -> skip it.
__global__ __launch_bounds__(256) void lstm_pre_kernel(
    const float* __restrict__ Gxp, const float* __restrict__ Gxq, // [T*64][640] x@Wih^T + (bih+bhh)
    const _Float16* __restrict__ Whh,                             // [640][160]
    const int* __restrict__ plens, const int* __restrict__ qlens,
    float* __restrict__ Hp, _Float16* __restrict__ Hp16,          // [Tp*64][320]
    float* __restrict__ Hq, _Float16* __restrict__ Hq16,          // [Tq*64][320]
    float* __restrict__ gate_ws)                                  // 4 * 64*640
{
  __shared__ _Float16 h16[64*160];
  const int blk = blockIdx.x;
  const int seq = blk >> 1, dir = blk & 1;
  const int T = seq ? TQq : TPp;
  const float* Gx = seq ? Gxq : Gxp;
  const int* lens = seq ? qlens : plens;
  float* Hout = seq ? Hq : Hp;
  _Float16* Hout16 = seq ? Hq16 : Hp16;
  float* G = gate_ws + (size_t)blk * 64*640;
  const int off = dir ? 150 : 0;
  const int tid = threadIdx.x, lane = tid & 31, wid = tid >> 5;

  for (int i = tid; i < 64*160; i += 256) h16[i] = (_Float16)0.0f;
  float creg[38];
#pragma unroll
  for (int r = 0; r < 38; ++r) creg[r] = 0.0f;
  __syncthreads();

  for (int s = 0; s < T; ++s){
    int t = dir ? (T-1-s) : s;
    // gates = Gx[t] + h @ Whh^T   (M=64, N covers 608>=600, K=160)
    for (int tile = wid; tile < 4*38; tile += 8){
      int tm = tile / 38, tn = tile - tm*38;
      int row0 = tm*16, col0 = tn*16;
      int n = lane & 15, hf = lane >> 4;
      v8f acc;
      const float* gx = Gx + ((size_t)t*64 + row0)*640 + col0 + n;
#pragma unroll
      for (int v = 0; v < 8; ++v) acc[v] = gx[(size_t)(v + 8*hf)*640];
      for (int k0 = 0; k0 < 160; k0 += 32){
        v16h a = frag_a(h16, 160, row0, k0, lane);
        v16h b = frag_b(Whh, 160, col0, k0, lane);
        WMMA(acc, a, b);
      }
      frag_store(G, 640, row0, col0, lane, acc);
    }
    __threadfence(); __syncthreads();
#pragma unroll 4
    for (int r = 0; r < 38; ++r){
      int e = tid + r*256;
      if (e < 64*150){
        int b = e / 150, j = e - b*150;
        float gi = G[b*640 + j];
        float gf = G[b*640 + 150 + j];
        float gg = G[b*640 + 300 + j];
        float mt = (t < lens[b]) ? 1.0f : 0.0f;
        float c2 = sigf(gf)*creg[r] + sigf(gi)*tanhf(gg);
        float h2 = tanhf(c2) * mt;
        c2 *= mt;
        creg[r] = c2;
        h16[b*160 + j] = (_Float16)h2;
        size_t o = ((size_t)t*64 + b)*320 + off + j;
        Hout[o] = h2;
        Hout16[o] = (_Float16)h2;
      }
    }
    __threadfence(); __syncthreads();
  }
}

// ---------------- persistent match-LSTM (2 blocks: fwd,bwd) ----------------
__global__ __launch_bounds__(512) void lstm_match_kernel(
    const float* __restrict__ Gm_hp,   // [Tp*64][640] = Hp @ m_Wih[:, :300]^T
    const float* __restrict__ app,     // [Tp*64][160] = Hp @ ap_W^T + ap_b
    const float* __restrict__ attq,    // [Tq*64][160] = Hq @ aq_W^T
    const float* __restrict__ Hq,      // [Tq*64][320] f32
    const _Float16* __restrict__ w_ah,    // [160][160]
    const _Float16* __restrict__ w_mWihR, // [640][320]  m_Wih[:, 300:600]
    const _Float16* __restrict__ w_mWhh,  // [640][160]
    const float* __restrict__ b_m,        // [640] = m_bih+m_bhh
    const float* __restrict__ al_W, const float* __restrict__ al_b,
    const int* __restrict__ plens,
    float* __restrict__ Hr, _Float16* __restrict__ Hr16, // [Tp*64][320]
    float* __restrict__ gate_ws)                         // 2 * 64*640
{
  extern __shared__ char smem[];
  _Float16* h16  = (_Float16*)smem;                          // 64*160
  _Float16* zw16 = (_Float16*)(smem + 64*160*2);             // 64*320 (w part of z, masked)
  float*    hh   = (float*)(smem + 64*160*2 + 64*320*2);     // 64*160
  float*    sc   = hh + 64*160;                              // 64*64 scores [tq*64+b]
  float*    alw  = sc + 64*64;                               // 160

  const int dir = blockIdx.x;
  const int offc = dir ? 150 : 0;
  float* G = gate_ws + (size_t)dir * 64*640;
  const int tid = threadIdx.x, lane = tid & 31, wid = tid >> 5;

  for (int i = tid; i < 64*160; i += 512) h16[i] = (_Float16)0.0f;
  for (int i = tid; i < 64*320; i += 512) zw16[i] = (_Float16)0.0f;
  for (int i = tid; i < 160; i += 512) alw[i] = (i < 150) ? al_W[i] : 0.0f;
  float creg[19];
#pragma unroll
  for (int r = 0; r < 19; ++r) creg[r] = 0.0f;
  __syncthreads();

  const float albv = al_b[0];
  for (int s = 0; s < TPp; ++s){
    int tt = dir ? (TPp-1-s) : s;
    // A: hh = h @ ah_W^T   (M=64,N=160,K=160)
    for (int tile = wid; tile < 4*10; tile += 16){
      int tm = tile / 10, tn = tile - tm*10;
      v8f acc = {};
      for (int k0 = 0; k0 < 160; k0 += 32){
        v16h a = frag_a(h16, 160, tm*16, k0, lane);
        v16h b = frag_b(w_ah, 160, tn*16, k0, lane);
        WMMA(acc, a, b);
      }
      frag_store(hh, 160, tm*16, tn*16, lane, acc);
    }
    __syncthreads();
    // B: scores[tq][b] = sum_j tanh(attq+app+hh)*al_W + al_b
    for (int o = tid; o < TQq*64; o += 512){
      int tq = o >> 6, b = o & 63;
      const float* aq_ = attq + ((size_t)tq*64 + b)*160;
      const float* ap_ = app  + ((size_t)tt*64 + b)*160;
      const float* hb_ = hh + b*160;
      float sum = 0.0f;
      for (int j = 0; j < 150; ++j) sum += tanhf(aq_[j] + ap_[j] + hb_[j]) * alw[j];
      sc[o] = sum + albv;
    }
    __syncthreads();
    // C: softmax over tq (per batch column)
    if (tid < 64){
      int b = tid;
      float mx = -3.0e38f;
      for (int tq = 0; tq < TQq; ++tq){ float v = sc[tq*64+b]; mx = v > mx ? v : mx; }
      float ssum = 0.0f;
      for (int tq = 0; tq < TQq; ++tq){ float e = expf(sc[tq*64+b]-mx); sc[tq*64+b] = e; ssum += e; }
      float inv = 1.0f/ssum;
      for (int tq = 0; tq < TQq; ++tq) sc[tq*64+b] *= inv;
    }
    __syncthreads();
    // D: w = sum_tq alpha*Hq ; store masked f16 (w-part of z)
    for (int o = tid; o < 64*300; o += 512){
      int b = o / 300, k = o - b*300;
      float sum = 0.0f;
      for (int tq = 0; tq < TQq; ++tq) sum += sc[tq*64+b]*Hq[((size_t)tq*64+b)*320 + k];
      float mt = (tt < plens[b]) ? 1.0f : 0.0f;
      zw16[b*320 + k] = (_Float16)(sum*mt);
    }
    __syncthreads();
    // E: gates = mt*Gm_hp[tt] + b_m + zw@mWihR^T + h@mWhh^T
    for (int tile = wid; tile < 4*38; tile += 16){
      int tm = tile / 38, tn = tile - tm*38;
      int row0 = tm*16, col0 = tn*16;
      int n = lane & 15, hf = lane >> 4;
      v8f acc;
#pragma unroll
      for (int v = 0; v < 8; ++v){
        int row = row0 + v + 8*hf;
        float mt = (tt < plens[row]) ? 1.0f : 0.0f;
        acc[v] = mt*Gm_hp[((size_t)tt*64 + row)*640 + col0 + n] + b_m[col0 + n];
      }
      for (int k0 = 0; k0 < 320; k0 += 32){
        v16h a = frag_a(zw16, 320, row0, k0, lane);
        v16h b = frag_b(w_mWihR, 320, col0, k0, lane);
        WMMA(acc, a, b);
      }
      for (int k0 = 0; k0 < 160; k0 += 32){
        v16h a = frag_a(h16, 160, row0, k0, lane);
        v16h b = frag_b(w_mWhh, 160, col0, k0, lane);
        WMMA(acc, a, b);
      }
      frag_store(G, 640, row0, col0, lane, acc);
    }
    __threadfence(); __syncthreads();
    // F: cell update
#pragma unroll 4
    for (int r = 0; r < 19; ++r){
      int e = tid + r*512;
      if (e < 64*150){
        int b = e / 150, j = e - b*150;
        float gi = G[b*640 + j];
        float gf = G[b*640 + 150 + j];
        float gg = G[b*640 + 300 + j];
        float go = G[b*640 + 450 + j];
        float mt = (tt < plens[b]) ? 1.0f : 0.0f;
        float c2 = sigf(gf)*creg[r] + sigf(gi)*tanhf(gg);
        float h2 = sigf(go)*tanhf(c2);
        h2 *= mt; c2 *= mt;
        creg[r] = c2;
        h16[b*160 + j] = (_Float16)h2;
        size_t o = ((size_t)tt*64 + b)*320 + offc + j;
        Hr[o] = h2;
        Hr16[o] = (_Float16)h2;
      }
    }
    __threadfence(); __syncthreads();
  }
}

// ---------------- question attention pool (alpha_q, wq) ----------------
__global__ __launch_bounds__(512) void attn_pool_q_kernel(
    const float* __restrict__ t1,   // [Tq*64][160] = Hq @ aqa_W^T
    const float* __restrict__ Hq,   // [Tq*64][320]
    const float* __restrict__ ala_W, const float* __restrict__ ala_b,
    float* __restrict__ wq, _Float16* __restrict__ wq16)  // [64][320]
{
  __shared__ float sc[64*64];
  const int tid = threadIdx.x;
  for (int o = tid; o < TQq*64; o += 512){
    int tq = o >> 6, b = o & 63;
    const float* p = t1 + ((size_t)tq*64 + b)*160;
    float sum = 0.0f;
    for (int j = 0; j < 150; ++j) sum += tanhf(p[j]) * ala_W[j];
    sc[o] = sum + ala_b[0];
  }
  __syncthreads();
  if (tid < 64){
    int b = tid;
    float mx = -3.0e38f;
    for (int tq = 0; tq < TQq; ++tq){ float v = sc[tq*64+b]; mx = v > mx ? v : mx; }
    float ssum = 0.0f;
    for (int tq = 0; tq < TQq; ++tq){ float e = expf(sc[tq*64+b]-mx); sc[tq*64+b] = e; ssum += e; }
    float inv = 1.0f/ssum;
    for (int tq = 0; tq < TQq; ++tq) sc[tq*64+b] *= inv;
  }
  __syncthreads();
  for (int o = tid; o < 64*320; o += 512){
    int b = o / 320, k = o - b*320;
    float val = 0.0f;
    if (k < 300){
      for (int tq = 0; tq < TQq; ++tq) val += sc[tq*64+b]*Hq[((size_t)tq*64+b)*320 + k];
    }
    wq[o] = val;
    wq16[o] = (_Float16)val;
  }
}

// ---------------- answer pointer (2 blocks: fwd,bwd branch) ----------------
__global__ __launch_bounds__(512) void answer_kernel(
    const float* __restrict__ amlA, const float* __restrict__ amlB, // [Tp*64][320]
    const float* __restrict__ wq, const _Float16* __restrict__ wq16,
    const float* __restrict__ Hr,                                   // [Tp*64][320]
    const _Float16* __restrict__ w_aans, const float* __restrict__ b_aans, // [320][320],[320]
    const _Float16* __restrict__ w_ansWih, const _Float16* __restrict__ w_ansWhh, // [1216][640],[1216][320]
    const float* __restrict__ b_ans,                                // [1216]
    const float* __restrict__ btW, const float* __restrict__ btb,
    const int* __restrict__ plens,
    float* __restrict__ out,        // [4][64][400]
    float* __restrict__ gate_ws,    // 2 * 64*1216
    float* __restrict__ beta_ws)    // 2 * 400*64
{
  extern __shared__ char smem[];
  _Float16* h16  = (_Float16*)smem;                         // 64*320
  _Float16* af16 = (_Float16*)(smem + 64*320*2);            // 64*640  (z = [wr, wq])
  float*    hh   = (float*)(smem + 64*320*2 + 64*640*2);    // 64*320

  const int br = blockIdx.x;
  const float* aml = br ? amlB : amlA;
  float* G = gate_ws + (size_t)br * 64*1216;
  float* beta = beta_ws + (size_t)br * 400*64;
  const int tid = threadIdx.x, lane = tid & 31, wid = tid >> 5;

  for (int i = tid; i < 64*320; i += 512) h16[i] = (_Float16)0.0f;
  for (int i = tid; i < 64*640; i += 512) af16[i] = (_Float16)0.0f;
  for (int o = tid; o < 64*300; o += 512){   // constant wq half of z
    int b = o / 300, j = o - b*300;
    af16[b*640 + 300 + j] = wq16[b*320 + j];
  }
  float creg[38];
#pragma unroll
  for (int r = 0; r < 38; ++r) creg[r] = 0.0f;
  __syncthreads();

  const float btbv = btb[0];
  for (int k = 0; k < 3; ++k){
    // hh = h @ aans_W^T + aans_b
    for (int tile = wid; tile < 4*20; tile += 16){
      int tm = tile / 20, tn = tile - tm*20;
      int n = lane & 15, hf = lane >> 4;
      v8f acc;
#pragma unroll
      for (int v = 0; v < 8; ++v) acc[v] = b_aans[tn*16 + n];
      for (int k0 = 0; k0 < 320; k0 += 32){
        v16h a = frag_a(h16, 320, tm*16, k0, lane);
        v16h b = frag_b(w_aans, 320, tn*16, k0, lane);
        WMMA(acc, a, b);
      }
      frag_store(hh, 320, tm*16, tn*16, lane, acc);
    }
    __syncthreads();
    // masked scores over passage
    for (int o = tid; o < TPp*64; o += 512){
      int t = o >> 6, b = o & 63;
      const float* am_ = aml + ((size_t)t*64 + b)*320;
      const float* hb_ = hh + b*320;
      const float* wq_ = wq + b*320;
      float sum = 0.0f;
      for (int j = 0; j < 300; ++j) sum += tanhf(am_[j] + hb_[j] + wq_[j]) * btW[j];
      sum += btbv;
      beta[o] = (t < plens[b]) ? sum : -1.0e30f;
    }
    __threadfence(); __syncthreads();
    // softmax over t (per batch column); masked entries underflow to 0
    if (tid < 64){
      int b = tid;
      float mx = -3.0e38f;
      for (int t = 0; t < TPp; ++t){ float v = beta[t*64+b]; mx = v > mx ? v : mx; }
      float ssum = 0.0f;
      for (int t = 0; t < TPp; ++t){ float e = expf(beta[t*64+b]-mx); beta[t*64+b] = e; ssum += e; }
      float inv = 1.0f/ssum;
      for (int t = 0; t < TPp; ++t){
        float mt = (t < plens[b]) ? 1.0f : 0.0f;
        beta[t*64+b] *= inv*mt;
      }
    }
    __threadfence(); __syncthreads();
    if (k > 0){
      int orow = br*2 + (k-1);
      for (int o = tid; o < TPp*64; o += 512){
        int t = o >> 6, b = o & 63;
        out[((size_t)orow*64 + b)*400 + t] = beta[o];
      }
    }
    // wr = sum_t beta*Hr ; write into z
    for (int o = tid; o < 64*300; o += 512){
      int b = o / 300, j = o - b*300;
      float sum = 0.0f;
      for (int t = 0; t < TPp; ++t) sum += beta[t*64+b]*Hr[((size_t)t*64+b)*320 + j];
      af16[b*640 + j] = (_Float16)sum;
    }
    __threadfence(); __syncthreads();
    // gates8 = af @ ans_Wih^T + (bih+bhh) + h @ ans_Whh^T   (N=1200 -> 75 tiles)
    for (int tile = wid; tile < 4*75; tile += 16){
      int tm = tile / 75, tn = tile - tm*75;
      int n = lane & 15, hf = lane >> 4;
      v8f acc;
#pragma unroll
      for (int v = 0; v < 8; ++v) acc[v] = b_ans[tn*16 + n];
      for (int k0 = 0; k0 < 640; k0 += 32){
        v16h a = frag_a(af16, 640, tm*16, k0, lane);
        v16h b = frag_b(w_ansWih, 640, tn*16, k0, lane);
        WMMA(acc, a, b);
      }
      for (int k0 = 0; k0 < 320; k0 += 32){
        v16h a = frag_a(h16, 320, tm*16, k0, lane);
        v16h b = frag_b(w_ansWhh, 320, tn*16, k0, lane);
        WMMA(acc, a, b);
      }
      frag_store(G, 1216, tm*16, tn*16, lane, acc);
    }
    __threadfence(); __syncthreads();
    // cell update (2H state)
#pragma unroll 4
    for (int r = 0; r < 38; ++r){
      int e = tid + r*512;
      if (e < 64*300){
        int b = e / 300, j = e - b*300;
        float gi = G[b*1216 + j];
        float gf = G[b*1216 + 300 + j];
        float gg = G[b*1216 + 600 + j];
        float go = G[b*1216 + 900 + j];
        float c2 = sigf(gf)*creg[r] + sigf(gi)*tanhf(gg);
        float h2 = sigf(go)*tanhf(c2);
        creg[r] = c2;
        h16[b*320 + j] = (_Float16)h2;
      }
    }
    __threadfence(); __syncthreads();
  }
}

// ---------------- host launcher ----------------
extern "C" void kernel_launch(void* const* d_in, const int* in_sizes, int n_in,
                              void* d_out, int out_size, void* d_ws, size_t ws_size,
                              hipStream_t stream)
{
  (void)in_sizes; (void)n_in; (void)out_size; (void)ws_size;
  const int*   pass_ids = (const int*)d_in[0];
  const int*   ques_ids = (const int*)d_in[1];
  const int*   plens    = (const int*)d_in[2];
  const int*   qlens    = (const int*)d_in[3];
  const float* emb      = (const float*)d_in[4];
  const float* pre_Wih  = (const float*)d_in[5];
  const float* pre_Whh  = (const float*)d_in[6];
  const float* pre_bih  = (const float*)d_in[7];
  const float* pre_bhh  = (const float*)d_in[8];
  const float* aq_W     = (const float*)d_in[9];
  const float* ap_W     = (const float*)d_in[10];
  const float* ap_b     = (const float*)d_in[11];
  const float* ah_W     = (const float*)d_in[12];
  const float* al_W     = (const float*)d_in[13];
  const float* al_b     = (const float*)d_in[14];
  const float* m_Wih    = (const float*)d_in[15];
  const float* m_Whh    = (const float*)d_in[16];
  const float* m_bih    = (const float*)d_in[17];
  const float* m_bhh    = (const float*)d_in[18];
  const float* aqa_W    = (const float*)d_in[19];
  const float* ala_W    = (const float*)d_in[20];
  const float* ala_b    = (const float*)d_in[21];
  const float* aml_W    = (const float*)d_in[22];
  const float* amlb_W   = (const float*)d_in[23];
  const float* aans_W   = (const float*)d_in[24];
  const float* aans_b   = (const float*)d_in[25];
  const float* bt_W     = (const float*)d_in[26];
  const float* bt_b     = (const float*)d_in[27];
  const float* ans_Wih  = (const float*)d_in[28];
  const float* ans_Whh  = (const float*)d_in[29];
  const float* ans_bih  = (const float*)d_in[30];
  const float* ans_bhh  = (const float*)d_in[31];
  float* out = (float*)d_out;

  // workspace carve-up
  char* base = (char*)d_ws; size_t off = 0;
  auto carve = [&](size_t bytes)->char*{
    char* p = base + off; off = (off + bytes + 255) & ~(size_t)255; return p;
  };
  const size_t RP = (size_t)TPp*64, RQ = (size_t)TQq*64;   // 25600, 3840
  _Float16* Xp_h   = (_Float16*)carve(RP*320*2);
  _Float16* Xq_h   = (_Float16*)carve(RQ*320*2);
  _Float16* Hp16   = (_Float16*)carve(RP*320*2);
  _Float16* Hq16   = (_Float16*)carve(RQ*320*2);
  _Float16* Hr16   = (_Float16*)carve(RP*320*2);
  _Float16* wq16   = (_Float16*)carve(64*320*2);
  _Float16* w_preWih = (_Float16*)carve(640*320*2);
  _Float16* w_preWhh = (_Float16*)carve(640*160*2);
  _Float16* w_aq     = (_Float16*)carve(160*320*2);
  _Float16* w_ap     = (_Float16*)carve(160*320*2);
  _Float16* w_ah     = (_Float16*)carve(160*160*2);
  _Float16* w_mWihL  = (_Float16*)carve(640*320*2);
  _Float16* w_mWihR  = (_Float16*)carve(640*320*2);
  _Float16* w_mWhh   = (_Float16*)carve(640*160*2);
  _Float16* w_aqa    = (_Float16*)carve(160*320*2);
  _Float16* w_aml    = (_Float16*)carve(320*320*2);
  _Float16* w_amlb   = (_Float16*)carve(320*320*2);
  _Float16* w_aans   = (_Float16*)carve(320*320*2);
  _Float16* w_ansWih = (_Float16*)carve((size_t)1216*640*2);
  _Float16* w_ansWhh = (_Float16*)carve((size_t)1216*320*2);
  float* Gxp   = (float*)carve(RP*640*4);
  float* Gxq   = (float*)carve(RQ*640*4);
  float* Hp    = (float*)carve(RP*320*4);
  float* Hq    = (float*)carve(RQ*320*4);
  float* Hr    = (float*)carve(RP*320*4);
  float* attq  = (float*)carve(RQ*160*4);
  float* app   = (float*)carve(RP*160*4);
  float* Gm_hp = (float*)carve(RP*640*4);
  float* amlF  = (float*)carve(RP*320*4);
  float* amlBk = (float*)carve(RP*320*4);
  float* t1    = (float*)carve(RQ*160*4);
  float* wq    = (float*)carve(64*320*4);
  float* b_pre = (float*)carve(640*4);
  float* b_ap  = (float*)carve(160*4);
  float* b_m   = (float*)carve(640*4);
  float* b_aans= (float*)carve(320*4);
  float* b_ans = (float*)carve(1216*4);
  float* g_pre = (float*)carve((size_t)4*64*640*4);
  float* g_mat = (float*)carve((size_t)2*64*640*4);
  float* g_ans = (float*)carve((size_t)2*64*1216*4);
  float* betaw = (float*)carve((size_t)2*400*64*4);

  auto blocks1 = [](size_t n){ return (unsigned)((n + 255)/256); };
  auto padw = [&](const float* s, _Float16* d, int N, int K, int lds, int koff, int Np, int Kp){
    pad_w16_kernel<<<blocks1((size_t)Np*Kp), 256, 0, stream>>>(s, d, N, K, lds, koff, Np, Kp);
  };
  auto padb = [&](const float* a, const float* b, float* d, int N, int Np){
    pad_bias_kernel<<<blocks1(Np), 256, 0, stream>>>(a, b, d, N, Np);
  };
  auto gemm = [&](const _Float16* A, int lda, const _Float16* W, int ldw,
                  const float* bias, float* C, int ldc, int Mt, int Nt, int K){
    size_t waves = (size_t)Mt*Nt;
    gemm_f16f32<<<blocks1(waves*32), 256, 0, stream>>>(A, lda, W, ldw, bias, C, ldc, Mt, Nt, K);
  };

  // 0) weight/bias prep
  padw(pre_Wih, w_preWih, 600, 300, 300,   0, 640, 320);
  padw(pre_Whh, w_preWhh, 600, 150, 150,   0, 640, 160);
  padw(aq_W,    w_aq,     150, 300, 300,   0, 160, 320);
  padw(ap_W,    w_ap,     150, 300, 300,   0, 160, 320);
  padw(ah_W,    w_ah,     150, 150, 150,   0, 160, 160);
  padw(m_Wih,   w_mWihL,  600, 300, 600,   0, 640, 320);
  padw(m_Wih,   w_mWihR,  600, 300, 600, 300, 640, 320);
  padw(m_Whh,   w_mWhh,   600, 150, 150,   0, 640, 160);
  padw(aqa_W,   w_aqa,    150, 300, 300,   0, 160, 320);
  padw(aml_W,   w_aml,    300, 300, 300,   0, 320, 320);
  padw(amlb_W,  w_amlb,   300, 300, 300,   0, 320, 320);
  padw(aans_W,  w_aans,   300, 300, 300,   0, 320, 320);
  padw(ans_Wih, w_ansWih, 1200, 600, 600,  0, 1216, 640);
  padw(ans_Whh, w_ansWhh, 1200, 300, 300,  0, 1216, 320);
  padb(pre_bih, pre_bhh, b_pre, 600, 640);
  padb(ap_b,    nullptr, b_ap,  150, 160);
  padb(m_bih,   m_bhh,   b_m,   600, 640);
  padb(aans_b,  nullptr, b_aans,300, 320);
  padb(ans_bih, ans_bhh, b_ans, 1200, 1216);

  // zero f16 activation buffers (pad cols feed WMMA A operands -> avoid NaN poison)
  zero_h16_kernel<<<blocks1(RP*320), 256, 0, stream>>>(Hp16, RP*320);
  zero_h16_kernel<<<blocks1(RQ*320), 256, 0, stream>>>(Hq16, RQ*320);
  zero_h16_kernel<<<blocks1(RP*320), 256, 0, stream>>>(Hr16, RP*320);

  // 1) embedding gather (f32 -> padded f16)
  gather_emb_kernel<<<blocks1(RP*320), 256, 0, stream>>>(pass_ids, emb, Xp_h, (int)RP);
  gather_emb_kernel<<<blocks1(RQ*320), 256, 0, stream>>>(ques_ids, emb, Xq_h, (int)RQ);

  // 2) big parallel GEMMs hoisted out of the recurrence
  gemm(Xp_h, 320, w_preWih, 320, b_pre, Gxp, 640, 1600, 40, 320);
  gemm(Xq_h, 320, w_preWih, 320, b_pre, Gxq, 640,  240, 40, 320);

  // 3) persistent bi-LSTM preprocess (4 independent sequences)
  lstm_pre_kernel<<<4, 256, 0, stream>>>(Gxp, Gxq, w_preWhh, plens, qlens,
                                         Hp, Hp16, Hq, Hq16, g_pre);

  // 4) match precomputes
  gemm(Hq16, 320, w_aq,    320, nullptr, attq,  160,  240, 10, 320);
  gemm(Hp16, 320, w_ap,    320, b_ap,    app,   160, 1600, 10, 320);
  gemm(Hp16, 320, w_mWihL, 320, nullptr, Gm_hp, 640, 1600, 40, 320);

  // 5) persistent match-LSTM (fwd + bwd)
  size_t smem_match = (size_t)64*160*2 + 64*320*2 + 64*160*4 + 64*64*4 + 160*4;
  lstm_match_kernel<<<2, 512, smem_match, stream>>>(
      Gm_hp, app, attq, Hq, w_ah, w_mWihR, w_mWhh, b_m, al_W, al_b, plens,
      Hr, Hr16, g_mat);

  // 6) answer-pointer precomputes
  gemm(Hr16, 320, w_aml,  320, nullptr, amlF,  320, 1600, 20, 320);
  gemm(Hr16, 320, w_amlb, 320, nullptr, amlBk, 320, 1600, 20, 320);
  gemm(Hq16, 320, w_aqa,  320, nullptr, t1,    160,  240, 10, 320);
  attn_pool_q_kernel<<<1, 512, 0, stream>>>(t1, Hq, ala_W, ala_b, wq, wq16);

  // 7) answer pointer, 3 steps, fwd/bwd branches
  size_t smem_ans = (size_t)64*320*2 + 64*640*2 + 64*320*4;
  answer_kernel<<<2, 512, smem_ans, stream>>>(
      amlF, amlBk, wq, wq16, Hr, w_aans, b_aans, w_ansWih, w_ansWhh, b_ans,
      bt_W, bt_b, plens, out, g_ans, betaw);
}
